// LangNN_62921270886462
// MI455X (gfx1250) — matrix operand order repaired and checked by reading, not compiled, not run on hardware
//
#include <hip/hip_runtime.h>

typedef __attribute__((ext_vector_type(16))) _Float16 v16h;
typedef __attribute__((ext_vector_type(8)))  _Float16 v8h;
typedef __attribute__((ext_vector_type(8)))  float    v8f;

#define HDIM 25
#define HP   32
#define NSTEPS 10
#define WAVES 2
#define BLOCK (WAVES * 32)
#define TILE_HALFS 512                 // one 32x16 f16 B-tile (or 16x32 staging buffer)
#define NTILES 8                       // 128 padded gate columns / 16
#define NMAT 4                         // enc-x, enc-sum, dec-x, dec-h
#define WT_HALFS ((NMAT * NTILES + 2) * TILE_HALFS)  // + 2 actor tiles
#define STG_SLOTS 11                   // 10 enc outputs + 1 scratch (state / dec-h)

union V16U { v16h v; v8h h[2]; };

// Branch-free transcendentals: native v_tanh_f32 on gfx1250 if the builtin
// exists, otherwise exp/rcp identity (still branch-free; no EXEC games).
__device__ __forceinline__ float tanh_fast(float x) {
#if __has_builtin(__builtin_amdgcn_tanhf)
  return __builtin_amdgcn_tanhf(x);
#elif __has_builtin(__builtin_amdgcn_tanh_f32)
  return __builtin_amdgcn_tanh_f32(x);
#else
  // tanh(x) = 2/(1+exp(-2x)) - 1
  float e = __expf(-2.0f * x);
  return __builtin_fmaf(2.0f, __builtin_amdgcn_rcpf(1.0f + e), -1.0f);
#endif
}
__device__ __forceinline__ float sigf(float x) {
  return __builtin_fmaf(0.5f, tanh_fast(0.5f * x), 0.5f);
}

// B operand (32x16 f16): lane = (k>>4)*16 + n, halfs within lane are K ascending.
__device__ __forceinline__ v16h load_b_tile(const _Float16* wts, int mt, int lane) {
  const _Float16* p = wts + mt * TILE_HALFS + lane * 16;
  V16U u;
  u.h[0] = *(const v8h*)(p);
  u.h[1] = *(const v8h*)(p + 8);
  return u.v;
}

// A operand (16x32 f16) from row-major f16 [16][32] staging:
// lanes 0-15 row m: K 0..15 ; lanes 16-31 row m: K 16..31.
__device__ __forceinline__ v16h load_a(const _Float16* stg, int lane) {
  const _Float16* rowp = stg + (lane & 15) * HP;
  const int o = (lane >> 4) * 16;
  V16U u;
  u.h[0] = *(const v8h*)(rowp + o);
  u.h[1] = *(const v8h*)(rowp + o + 8);
  return u.v;
}

__device__ __forceinline__ v8f wmma_f16(v16h a, v16h b, v8f c) {
  return __builtin_amdgcn_wmma_f32_16x16x32_f16(false, a, false, b, (short)0, c,
                                                false, false);
}
__device__ __forceinline__ v8f wmma_z(v16h a, v16h b) {
  v8f z = {};                           // folds to inline SRC2=0
  return __builtin_amdgcn_wmma_f32_16x16x32_f16(false, a, false, b, (short)0, z,
                                                false, false);
}

// acc tiles: i=(0,1) f=(2,3) g=(4,5) o=(6,7); same lane slot = same unit column.
// Per-column gate biases added here (exact f32), instead of pre-splatting into C.
__device__ __forceinline__ void lstm_elem(const v8f acc[8], const float bias[8],
                                          v8f c[2], v8f h[2]) {
  #pragma unroll
  for (int p = 0; p < 2; ++p) {
    #pragma unroll
    for (int j = 0; j < 8; ++j) {
      float iv = acc[0 + p][j] + bias[0 + p];
      float fv = acc[2 + p][j] + bias[2 + p];
      float gv = acc[4 + p][j] + bias[4 + p];
      float ov = acc[6 + p][j] + bias[6 + p];
      float cn = sigf(fv) * c[p][j] + sigf(iv) * tanh_fast(gv);
      c[p][j] = cn;
      h[p][j] = sigf(ov) * tanh_fast(cn);
    }
  }
}

// C-layout f32 (row = v + 8*(lane>>4), col = lane&15 [+16]) -> row-major f16 staging
__device__ __forceinline__ void store_h_stg(_Float16* stg, const v8f h[2], int lane) {
  const int rb = (lane >> 4) * 8, cl = lane & 15;
  #pragma unroll
  for (int v = 0; v < 8; ++v) {
    stg[(rb + v) * HP + cl]      = (_Float16)h[0][v];
    stg[(rb + v) * HP + 16 + cl] = (_Float16)h[1][v];
  }
}

__device__ __forceinline__ void store_h_glob(float* dst, const v8f h[2], int lane) {
  const int rb = (lane >> 4) * 8, cl = lane & 15;
  #pragma unroll
  for (int v = 0; v < 8; ++v) {
    dst[(rb + v) * HDIM + cl] = h[0][v];
    if (cl < HDIM - 16) dst[(rb + v) * HDIM + 16 + cl] = h[1][v];
  }
}

__global__ __launch_bounds__(BLOCK) void langnn_lstm_wmma(
    const float* __restrict__ state,
    const float* __restrict__ Wih_enc, const float* __restrict__ Whh_enc,
    const float* __restrict__ bih_enc, const float* __restrict__ bhh_enc,
    const float* __restrict__ Wih_dec, const float* __restrict__ Whh_dec,
    const float* __restrict__ bih_dec, const float* __restrict__ bhh_dec,
    const float* __restrict__ Wact, const float* __restrict__ bact,
    float* __restrict__ out_act, float* __restrict__ out_enc, int B)
{
  __shared__ __align__(16) _Float16 wts[WT_HALFS];
  __shared__ __align__(16) _Float16 stg[WAVES][STG_SLOTS * TILE_HALFS];

  const int tid = threadIdx.x;

  // ---- pack weights (f32 global -> f16 LDS, B-operand layout, gate-padded) ----
  for (int idx = tid; idx < WT_HALFS; idx += BLOCK) {
    const int mt = idx / TILE_HALFS;
    const int q  = idx % TILE_HALFS;
    const int ln = q >> 4;                     // lane slot within tile
    const int hh = q & 15;                     // half within lane
    const int k  = ((ln >> 4) << 4) + hh;      // K index 0..31
    float val = 0.0f;
    if (mt < NMAT * NTILES) {
      const int m = mt >> 3;
      const int n = (mt & 7) * 16 + (ln & 15); // padded gate column 0..127
      const int gate = n >> 5, unit = n & 31;
      if (k < HDIM && unit < HDIM) {
        const int wr = (gate * HDIM + unit) * HDIM + k;
        if      (m == 0) val = Wih_enc[wr];                 // encoder step 1 (x)
        else if (m == 1) val = Wih_enc[wr] + Whh_enc[wr];   // encoder steps 2..10
        else if (m == 2) val = Wih_dec[wr];                 // decoder x
        else             val = Whh_dec[wr];                 // decoder h
      }
    } else {
      const int n = (mt - NMAT * NTILES) * 16 + (ln & 15);  // actor column 0..31
      if (k < HDIM && n < 18) val = Wact[n * HDIM + k];
    }
    wts[idx] = (_Float16)val;
  }
  __syncthreads();

  const int lane = tid & 31;
  const int wave = tid >> 5;
  const long long rowBase = ((long long)blockIdx.x * WAVES + wave) * 16;
  if (rowBase >= B) return;                    // uniform per wave: EXEC stays all-1s
  _Float16* mystg = stg[wave];
  _Float16* s10 = mystg + 10 * TILE_HALFS;

  // per-lane gate biases (column-dependent only)
  float encB[8], decB[8];
  #pragma unroll
  for (int t = 0; t < 8; ++t) {
    const int g = t >> 1, u = ((t & 1) << 4) + (lane & 15);
    float eb = 0.0f, db = 0.0f;
    if (u < HDIM) {
      eb = bih_enc[g * HDIM + u] + bhh_enc[g * HDIM + u];
      db = bih_dec[g * HDIM + u] + bhh_dec[g * HDIM + u];
    }
    encB[t] = eb; decB[t] = db;
  }

  // ---- stage the 16x25 state tile (contiguous 400 f32) into slot 10, zero-padded
  {
    v8h zh;
    #pragma unroll
    for (int j = 0; j < 8; ++j) zh[j] = (_Float16)0.0f;
    *(v8h*)(s10 + lane * 16) = zh;
    *(v8h*)(s10 + lane * 16 + 8) = zh;
    const float* sp = state + rowBase * HDIM;
    for (int idx = lane; idx < 16 * HDIM; idx += 32)
      s10[(idx / HDIM) * HP + (idx % HDIM)] = (_Float16)sp[idx];
  }

  v8f c[2], h[2], acc[8];
  #pragma unroll
  for (int p = 0; p < 2; ++p)
    #pragma unroll
    for (int j = 0; j < 8; ++j) c[p][j] = 0.0f;

  // ---- encoder step 1: gates = state @ Wih_enc^T + b  (h=0) ----
  {
    v16h xA = load_a(s10, lane);
    #pragma unroll
    for (int t = 0; t < 8; ++t)
      acc[t] = wmma_z(xA, load_b_tile(wts, t, lane));
    lstm_elem(acc, encB, c, h);
    store_h_glob(out_enc + rowBase * HDIM, h, lane);
    store_h_stg(mystg, h, lane);
  }
  // ---- encoder steps 2..10: gates = h @ (Wih_enc + Whh_enc)^T + b ----
  for (int s = 1; s < NSTEPS; ++s) {
    v16h hA = load_a(mystg + (s - 1) * TILE_HALFS, lane);
    #pragma unroll
    for (int t = 0; t < 8; ++t)
      acc[t] = wmma_z(hA, load_b_tile(wts, NTILES + t, lane));
    lstm_elem(acc, encB, c, h);
    store_h_glob(out_enc + ((long long)s * B + rowBase) * HDIM, h, lane);
    store_h_stg(mystg + s * TILE_HALFS, h, lane);
  }

  // ---- decoder: consumes staged encoder outputs ----
  #pragma unroll
  for (int p = 0; p < 2; ++p)
    #pragma unroll
    for (int j = 0; j < 8; ++j) c[p][j] = 0.0f;
  {
    v8h zh;
    #pragma unroll
    for (int j = 0; j < 8; ++j) zh[j] = (_Float16)0.0f;
    *(v8h*)(s10 + lane * 16) = zh;           // dec h starts at zero
    *(v8h*)(s10 + lane * 16 + 8) = zh;
  }
  for (int s = 0; s < NSTEPS; ++s) {
    v16h xA = load_a(mystg + s * TILE_HALFS, lane);
    v16h hA = load_a(s10, lane);
    #pragma unroll
    for (int t = 0; t < 8; ++t) {
      acc[t] = wmma_z(xA, load_b_tile(wts, 2 * NTILES + t, lane));
      acc[t] = wmma_f16(hA, load_b_tile(wts, 3 * NTILES + t, lane), acc[t]);
    }
    lstm_elem(acc, decB, c, h);
    store_h_stg(s10, h, lane);               // becomes next step's hA (and actor input)
  }

  // ---- actor head: out = h_dec @ Wact^T + b_act  (18 cols -> 2 tiles) ----
  {
    v16h hA = load_a(s10, lane);
    const int cl = lane & 15;
    const float b0 = bact[cl];
    const float b1 = (cl < 2) ? bact[16 + cl] : 0.0f;
    v8f a0 = wmma_z(hA, load_b_tile(wts, NMAT * NTILES,     lane));
    v8f a1 = wmma_z(hA, load_b_tile(wts, NMAT * NTILES + 1, lane));
    float* oa = out_act + rowBase * 18;
    const int rb = (lane >> 4) * 8;
    #pragma unroll
    for (int v = 0; v < 8; ++v) {
      oa[(rb + v) * 18 + cl] = a0[v] + b0;
      if (cl < 2) oa[(rb + v) * 18 + 16 + cl] = a1[v] + b1;
    }
  }
}

extern "C" void kernel_launch(void* const* d_in, const int* in_sizes, int n_in,
                              void* d_out, int out_size, void* d_ws, size_t ws_size,
                              hipStream_t stream) {
  (void)n_in; (void)out_size; (void)d_ws; (void)ws_size;
  const float* state   = (const float*)d_in[0];
  const float* Wih_enc = (const float*)d_in[1];
  const float* Whh_enc = (const float*)d_in[2];
  const float* bih_enc = (const float*)d_in[3];
  const float* bhh_enc = (const float*)d_in[4];
  const float* Wih_dec = (const float*)d_in[5];
  const float* Whh_dec = (const float*)d_in[6];
  const float* bih_dec = (const float*)d_in[7];
  const float* bhh_dec = (const float*)d_in[8];
  const float* Wact    = (const float*)d_in[9];
  const float* bact    = (const float*)d_in[10];
  const int B = in_sizes[0] / HDIM;            // 131072
  float* out_act = (float*)d_out;              // [B,18]
  float* out_enc = out_act + (size_t)B * 18;   // [10,B,25]
  const int rowsPerBlock = WAVES * 16;
  dim3 grid((B + rowsPerBlock - 1) / rowsPerBlock), block(BLOCK);
  hipLaunchKernelGGL(langnn_lstm_wmma, grid, block, 0, stream,
                     state, Wih_enc, Whh_enc, bih_enc, bhh_enc,
                     Wih_dec, Whh_dec, bih_dec, bhh_dec, Wact, bact,
                     out_act, out_enc, B);
}
